// QuantumAttention_1889785610324
// MI455X (gfx1250) — matrix operand order repaired and checked by reading, not compiled, not run on hardware
//
#include <hip/hip_runtime.h>

// B=4, T=2048, D=1024
#define B_DIM 4
#define T_DIM 2048
#define D_DIM 1024

typedef float f4 __attribute__((ext_vector_type(4)));
typedef __attribute__((ext_vector_type(16))) __bf16 v16bf;
typedef __attribute__((ext_vector_type(8)))  float  v8f;

// Workspace layout (float offsets)
#define WS_WBAR   0        // [1024]  column means of Wk
#define WS_BBAR   1024     // [1]     mean of bk
#define WS_KMEAN  2048     // [B*T]
#define WS_W      10240    // [B*T]   softmax weights
#define WS_XWPART 18432    // [4][B][1024] t-sliced partial weighted sums of x
#define WS_XWPAD  34816    // [16][1024]   WMMA A operand (rows 4..15 zero)
#define WS_OROW   51200    // [B][1024]    out row per batch

// ---------------------------------------------------------------------------
// Kernel 1: wbar[d] = (1/D) * sum_e Wk[e,d]   ;   bbar = (1/D) * sum_e bk[e]
__global__ void qa_prep(const float* __restrict__ Wk,
                        const float* __restrict__ bk,
                        float* ws) {
  const int tid = threadIdx.x;
  if (blockIdx.x < 4) {
    const int d = blockIdx.x * 256 + tid;
    float s = 0.f;
    for (int e = 0; e < D_DIM; ++e) s += Wk[(size_t)e * D_DIM + d]; // coalesced
    ws[WS_WBAR + d] = s * (1.0f / D_DIM);
  } else {
    __shared__ float red[256];
    float s = bk[tid] + bk[tid + 256] + bk[tid + 512] + bk[tid + 768];
    red[tid] = s; __syncthreads();
    for (int off = 128; off > 0; off >>= 1) {
      if (tid < off) red[tid] += red[tid + off];
      __syncthreads();
    }
    if (tid == 0) ws[WS_BBAR] = red[0] * (1.0f / D_DIM);
  }
}

// ---------------------------------------------------------------------------
// Kernel 2: kmean[b,t] = x[b,t,:] . wbar + bbar    (one wave32 per row)
__global__ void qa_kmean(const float* __restrict__ x, float* ws) {
  const int lane = threadIdx.x & 31;
  const int wave = threadIdx.x >> 5;
  const int row  = blockIdx.x * 8 + wave;                // 0 .. B*T-1
  const f4* xr = (const f4*)(x + (size_t)row * D_DIM);
  const f4* wb = (const f4*)(ws + WS_WBAR);
  float acc = 0.f;
  #pragma unroll
  for (int i = 0; i < 8; ++i) {                          // 256 float4 per row
    f4 xv = xr[i * 32 + lane];
    f4 wv = wb[i * 32 + lane];
    acc = fmaf(xv[0], wv[0], acc);
    acc = fmaf(xv[1], wv[1], acc);
    acc = fmaf(xv[2], wv[2], acc);
    acc = fmaf(xv[3], wv[3], acc);
  }
  #pragma unroll
  for (int off = 16; off > 0; off >>= 1) acc += __shfl_xor(acc, off, 32);
  if (lane == 0) ws[WS_KMEAN + row] = acc + ws[WS_BBAR];
}

// ---------------------------------------------------------------------------
// Kernel 3: weights[b,:] = softmax_t( cos(kmean[b,:]) )   (one block per b)
__global__ void qa_softmax(float* ws) {
  __shared__ float red[256];
  const int b = blockIdx.x, tid = threadIdx.x;
  const float* km = ws + WS_KMEAN + b * T_DIM;
  float sc[8]; float mx = -3.402823e38f;
  #pragma unroll
  for (int i = 0; i < 8; ++i) { sc[i] = cosf(km[i * 256 + tid]); mx = fmaxf(mx, sc[i]); }
  red[tid] = mx; __syncthreads();
  for (int off = 128; off > 0; off >>= 1) {
    if (tid < off) red[tid] = fmaxf(red[tid], red[tid + off]);
    __syncthreads();
  }
  mx = red[0]; __syncthreads();
  float ex[8]; float sum = 0.f;
  #pragma unroll
  for (int i = 0; i < 8; ++i) { ex[i] = expf(sc[i] - mx); sum += ex[i]; }
  red[tid] = sum; __syncthreads();
  for (int off = 128; off > 0; off >>= 1) {
    if (tid < off) red[tid] += red[tid + off];
    __syncthreads();
  }
  const float inv = 1.0f / red[0];
  #pragma unroll
  for (int i = 0; i < 8; ++i) ws[WS_W + b * T_DIM + i * 256 + tid] = ex[i] * inv;
}

// ---------------------------------------------------------------------------
// Kernel 4: xw_part[s][b][c] = sum_{t in slice s} w[b,t] * x[b,t,c]
// grid (colchunk=4, b=4, slice=4); second pass over x hits L2 (33.5MB < 192MB)
__global__ void qa_xw_part(const float* __restrict__ x, float* ws) {
  const int col = blockIdx.x * 256 + threadIdx.x;
  const int b = blockIdx.y, s = blockIdx.z;
  const float* wrow = ws + WS_W + b * T_DIM + s * 512;
  const float* xb   = x + (size_t)(b * T_DIM + s * 512) * D_DIM + col;
  float acc = 0.f;
  for (int t = 0; t < 512; ++t)
    acc = fmaf(wrow[t], xb[(size_t)t * D_DIM], acc);   // wrow[t] is a broadcast load
  ws[WS_XWPART + (s * 4 + b) * D_DIM + col] = acc;
}

// Kernel 5: reduce 4 slices into the padded 16x1024 WMMA A operand (rows>=4 zero)
__global__ void qa_xw_pad(float* ws) {
  const int idx = blockIdx.x * 256 + threadIdx.x;      // 0..16383
  const int m = idx >> 10, c = idx & 1023;
  float v = 0.f;
  if (m < B_DIM) {
    #pragma unroll
    for (int s = 0; s < 4; ++s) v += ws[WS_XWPART + (s * 4 + m) * D_DIM + c];
  }
  ws[WS_XWPAD + idx] = v;
}

// ---------------------------------------------------------------------------
// Kernel 6: out_row[16,1024] = xwPad @ Wv^T + bv via v_wmma_f32_16x16x32_bf16
// bf16x3 split (hi*hi + hi*lo + lo*hi, f32 accumulate) for ~fp32 accuracy.
// One wave per 16-column tile of d; 64 tiles; K-loop of 32 steps.
__global__ void __launch_bounds__(32) qa_wmma(const float* __restrict__ Wv,
                                              const float* __restrict__ bv,
                                              float* ws) {
  const int lane  = threadIdx.x & 31;
  const int hi    = lane >> 4;       // lane half
  const int ln    = lane & 15;       // M for A, N for B/C
  const int ntile = blockIdx.x;      // 0..63
  const float* xwPad = ws + WS_XWPAD;
  v8f acc = {0.f, 0.f, 0.f, 0.f, 0.f, 0.f, 0.f, 0.f};

  for (int kb = 0; kb < D_DIM; kb += 32) {
    v16bf a_hi, a_lo, b_hi, b_lo;
    #pragma unroll
    for (int v = 0; v < 8; ++v) {
      #pragma unroll
      for (int h = 0; h < 2; ++h) {
        const int e = 2 * v + h;
        // A (16x32 bf16): lanes0-15 hold K {0..7,16..23}; lanes16-31 {8..15,24..31}
        const int ka = ((v < 4) ? 2 * v : 2 * v + 8) + 8 * hi + h;
        const float av = xwPad[ln * D_DIM + kb + ka];
        const __bf16 ah = (__bf16)av;
        a_hi[e] = ah;
        a_lo[e] = (__bf16)(av - (float)ah);
        // B (32x16 bf16): lanes0-15 hold K {0..7,16..23}; lanes16-31 {8..15,24..31}
        const int kx = ((v < 4) ? 2 * v : 2 * v + 8) + 8 * hi + h;
        const float bb = Wv[(size_t)(ntile * 16 + ln) * D_DIM + kb + kx];
        const __bf16 bh = (__bf16)bb;
        b_hi[e] = bh;
        b_lo[e] = (__bf16)(bb - (float)bh);
      }
    }
    acc = __builtin_amdgcn_wmma_f32_16x16x32_bf16(false, a_hi, false, b_hi, (short)0, acc, false, false);
    acc = __builtin_amdgcn_wmma_f32_16x16x32_bf16(false, a_hi, false, b_lo, (short)0, acc, false, false);
    acc = __builtin_amdgcn_wmma_f32_16x16x32_bf16(false, a_lo, false, b_hi, (short)0, acc, false, false);
  }
  // C/D f32 16x16: lanes0-15 N=lane, VGPR r -> M=r; only M<4 are real batches.
  if (hi == 0) {
    const int dcol = ntile * 16 + ln;
    #pragma unroll
    for (int r = 0; r < B_DIM; ++r)
      ws[WS_OROW + r * D_DIM + dcol] = acc[r] + bv[dcol];
  }
}

// ---------------------------------------------------------------------------
// Kernel 7: out[b,t,:] = out_row[b,:]  (33.5 MB, non-temporal float4 stores)
__global__ void qa_broadcast(const float* __restrict__ ws, float* __restrict__ out) {
  const size_t idx4 = (size_t)blockIdx.x * 256 + threadIdx.x;
  const size_t elem = idx4 * 4;
  const int b = (int)(elem >> 21);          // T*D = 2^21
  const int d = (int)(elem & (D_DIM - 1));
  f4 v = *(const f4*)(ws + WS_OROW + b * D_DIM + d);
  __builtin_nontemporal_store(v, (f4*)(out + elem));
}

// ---------------------------------------------------------------------------
extern "C" void kernel_launch(void* const* d_in, const int* in_sizes, int n_in,
                              void* d_out, int out_size, void* d_ws, size_t ws_size,
                              hipStream_t stream) {
  const float* x  = (const float*)d_in[0];
  // d_in[1]=Wq, d_in[2]=bq are dead code in the reference.
  const float* Wk = (const float*)d_in[3];
  const float* bk = (const float*)d_in[4];
  const float* Wv = (const float*)d_in[5];
  const float* bv = (const float*)d_in[6];
  float* out = (float*)d_out;
  float* ws  = (float*)d_ws;
  (void)in_sizes; (void)n_in; (void)out_size; (void)ws_size;

  qa_prep     <<<5,    256, 0, stream>>>(Wk, bk, ws);
  qa_kmean    <<<1024, 256, 0, stream>>>(x, ws);
  qa_softmax  <<<4,    256, 0, stream>>>(ws);
  qa_xw_part  <<<dim3(4, 4, 4), 256, 0, stream>>>(x, ws);
  qa_xw_pad   <<<64,   256, 0, stream>>>(ws);
  qa_wmma     <<<64,   32,  0, stream>>>(Wv, bv, ws);
  qa_broadcast<<<8192, 256, 0, stream>>>(ws, out);
}
